// Attention_11407433138648
// MI455X (gfx1250) — matrix operand order repaired and checked by reading, not compiled, not run on hardware
//
#include <hip/hip_runtime.h>
#include <hip/hip_bf16.h>
#include <math.h>

// Problem constants (reference: B=2, I=2048, M=2048, D=1024, H=16, Dh=64)
#define BB 2
#define SI 2048
#define SM 2048
#define DD 1024
#define HH 16
#define DH 64

// LDS row stride for the 16 x 2048 fp32 logits block (+4 pad: kills bank
// conflicts on row-strided fragment reads, keeps 16B alignment per row)
#define LDS_STRIDE 2052

typedef __bf16 bf16_t;
typedef bf16_t v16bf __attribute__((ext_vector_type(16)));
typedef float  v8f   __attribute__((ext_vector_type(8)));

// ---------------------------------------------------------------------------
// WMMA fragment helpers (wave32, v_wmma_f32_16x16x32_bf16)
//
// A (16x32, bf16, row-major source, ld = lda):
//   lanes 0-15 : row = row0+lane,     slots 0..7 -> k0+0..7,  slots 8..15 -> k0+16..23
//   lanes 16-31: row = row0+lane-16,  slots 0..7 -> k0+8..15, slots 8..15 -> k0+24..31
// B (32x16) loaded from a TRANSPOSED source Bt[n][k] (ld = ldb):
//   lane n holds column col0+(lane&15); slots 0..7 -> k0+8*(lane>>4)+0..7, slots 8..15 -> k0+16+8*(lane>>4)+0..7
// C/D (16x16 f32): VGPR r, lanes 0-15 -> (row0+r, col0+lane); lanes 16-31 -> (+8)
// ---------------------------------------------------------------------------

__device__ inline v16bf load_A_frag_bf16(const bf16_t* __restrict__ A, int lda,
                                         int row0, int k0) {
  const int lane = threadIdx.x & 31;
  const int half = lane >> 4;
  const bf16_t* p = A + (size_t)(row0 + (lane & 15)) * lda + k0 + half * 8;
  v16bf a;
#pragma unroll
  for (int e = 0; e < 8; ++e) a[e] = p[e];
#pragma unroll
  for (int e = 0; e < 8; ++e) a[8 + e] = p[16 + e];
  return a;
}

// A-fragment sourced from the fp32 logits block in LDS (convert to bf16)
__device__ inline v16bf load_A_frag_lds(const float* __restrict__ s, int k0) {
  const int lane = threadIdx.x & 31;
  const int half = lane >> 4;
  const float* p = s + (lane & 15) * LDS_STRIDE + k0 + half * 8;
  v16bf a;
#pragma unroll
  for (int e = 0; e < 8; ++e) a[e] = (bf16_t)p[e];
#pragma unroll
  for (int e = 0; e < 8; ++e) a[8 + e] = (bf16_t)p[16 + e];
  return a;
}

__device__ inline v16bf load_Bt_frag(const bf16_t* __restrict__ Bt, size_t ldb,
                                     int col0, int k0) {
  const int lane = threadIdx.x & 31;
  const int half = lane >> 4;
  const bf16_t* p = Bt + (size_t)(col0 + (lane & 15)) * ldb + k0 + half * 8;
  v16bf b;
#pragma unroll
  for (int e = 0; e < 8; ++e) b[e] = p[e];
#pragma unroll
  for (int e = 0; e < 8; ++e) b[8 + e] = p[16 + e];
  return b;
}

__device__ inline v8f wmma_bf16(v16bf a, v16bf b, v8f c) {
  return __builtin_amdgcn_wmma_f32_16x16x32_bf16(false, a, false, b,
                                                 (short)0, c, false, false);
}

// ---------------------------------------------------------------------------
// Cast / transpose kernels
// ---------------------------------------------------------------------------
__global__ void cast_bf16(const float* __restrict__ src, bf16_t* __restrict__ dst, int n) {
  int i = blockIdx.x * blockDim.x + threadIdx.x;
  if (i < n) dst[i] = (bf16_t)src[i];
}

// Wt[n][k] = W[k][n] * scale  (so GEMM B-fragments are contiguous per lane)
__global__ void cast_w_trans(const float* __restrict__ W, bf16_t* __restrict__ Wt,
                             float scale) {
  int i = blockIdx.x * blockDim.x + threadIdx.x;  // i < DD*DD
  int n = i / DD, k = i % DD;
  Wt[(size_t)n * DD + k] = (bf16_t)(W[(size_t)k * DD + n] * scale);
}

// ---------------------------------------------------------------------------
// bf16 GEMM: C[row, col] = sum_k A[row,k] * Bt[col,k]
// Each wave computes a 16x64 strip (4 accumulators) so the A-fragment is
// reused 4x per K-step. 4 waves/block -> 16 x 256 per block.
// TRANS_OUT stores C transposed per batch: C[(b*N + col)*mseq + row%rows_per_batch]
// ---------------------------------------------------------------------------
template <bool TRANS_OUT, typename OutT>
__global__ void gemm_bf16(const bf16_t* __restrict__ A, const bf16_t* __restrict__ Bt,
                          OutT* __restrict__ C, int Kdim, int lda, int ldb, int Ncols,
                          int rows_per_batch, int mseq) {
  const int wave = threadIdx.x >> 5;
  const int row0 = blockIdx.x * 16;
  const int colbase = (blockIdx.y * 4 + wave) * 64;
  v8f acc[4] = {};
  for (int kk = 0; kk < Kdim; kk += 32) {
    v16bf a = load_A_frag_bf16(A, lda, row0, kk);
#pragma unroll
    for (int c = 0; c < 4; ++c) {
      v16bf b = load_Bt_frag(Bt, (size_t)ldb, colbase + c * 16, kk);
      acc[c] = wmma_bf16(a, b, acc[c]);
    }
  }
  const int lane = threadIdx.x & 31;
  const int half = lane >> 4;
#pragma unroll
  for (int c = 0; c < 4; ++c) {
    const int col = colbase + c * 16 + (lane & 15);
#pragma unroll
    for (int r = 0; r < 8; ++r) {
      int row = row0 + r + 8 * half;
      if constexpr (TRANS_OUT) {
        int b_ = row / rows_per_batch;
        int m_ = row % rows_per_batch;
        C[((size_t)b_ * Ncols + col) * (size_t)mseq + m_] = (OutT)acc[c][r];
      } else {
        C[(size_t)row * Ncols + col] = (OutT)acc[c][r];
      }
    }
  }
}

// ---------------------------------------------------------------------------
// Fused attention core: one workgroup (4 waves) per (b, h, 16-row i-block).
//   Phase 1: logits = Q K^T + bias  -> LDS (16 x 2048 fp32, 128 KB)
//   Phase 2: row softmax in LDS; stream align to HBM once via the async
//            LDS->global DMA path (global_store_async_from_lds_b128, ASYNCcnt)
//   Phase 3: ctx = align @ V from LDS-resident align (ds_load + cvt -> wmma)
// ---------------------------------------------------------------------------
__global__ void fused_attn_kernel(const bf16_t* __restrict__ Q,
                                  const bf16_t* __restrict__ Kb,
                                  const bf16_t* __restrict__ Vt,
                                  const float* __restrict__ bias,
                                  float* __restrict__ align,
                                  bf16_t* __restrict__ ctx) {
  extern __shared__ float smem[];            // [16][LDS_STRIDE] logits block
  float* red  = smem + 16 * LDS_STRIDE;      // [128] reduction scratch
  float* rowv = red + 128;                   // [16] per-row max / 1/sum

  const int tid  = threadIdx.x;
  const int wave = tid >> 5;
  const int lane = tid & 31;
  const int half = lane >> 4;

  const int i0 = blockIdx.x * 16;
  const int bh = blockIdx.z;
  const int b = bh / HH, h = bh % HH;

  const bf16_t* Qb_ = Q + (size_t)b * SI * DD;
  const bf16_t* Kt_ = Kb + (size_t)b * SM * DD;   // K row-major == (K^T) transposed
  const bf16_t* Vt_ = Vt + ((size_t)b * DD + h * DH) * SM;

  // ---- Phase 1: logits -> LDS ------------------------------------------------
  // Q fragments loaded ONCE and reused across all 32 m-tiles handled per wave.
  const v16bf qa0 = load_A_frag_bf16(Qb_, DD, i0, h * DH + 0);
  const v16bf qa1 = load_A_frag_bf16(Qb_, DD, i0, h * DH + 32);
  for (int t = wave; t < SM / 16; t += 4) {
    const int m0 = t * 16;
    v8f acc = {};
    acc = wmma_bf16(qa0, load_Bt_frag(Kt_, DD, m0, h * DH + 0), acc);
    acc = wmma_bf16(qa1, load_Bt_frag(Kt_, DD, m0, h * DH + 32), acc);
    const int m = m0 + (lane & 15);
#pragma unroll
    for (int r = 0; r < 8; ++r) {
      const int rr = r + 8 * half;
      smem[rr * LDS_STRIDE + m] =
          acc[r] + bias[((size_t)b * SI + i0 + rr) * SM + m];
    }
  }
  __syncthreads();

  // ---- Phase 2: softmax over each of the 16 rows (M = 2048) -----------------
  {
    const int r = tid >> 3;        // 8 threads per row
    const int sub = tid & 7;
    const float* prow = smem + r * LDS_STRIDE;

    float lmax = -INFINITY;
    for (int j = sub; j < SM; j += 8) lmax = fmaxf(lmax, prow[j]);
    red[tid] = lmax;
    __syncthreads();
    if (tid < 16) {
      float m0 = red[tid * 8];
#pragma unroll
      for (int j = 1; j < 8; ++j) m0 = fmaxf(m0, red[tid * 8 + j]);
      rowv[tid] = m0;
    }
    __syncthreads();

    float* prw = smem + r * LDS_STRIDE;
    const float mx = rowv[r];
    float lsum = 0.0f;
    for (int j = sub; j < SM; j += 8) {
      float e = __expf(prw[j] - mx);
      prw[j] = e;
      lsum += e;
    }
    __syncthreads();           // rowv reads done before overwrite below
    red[tid] = lsum;
    __syncthreads();
    if (tid < 16) {
      float s0 = 0.0f;
#pragma unroll
      for (int j = 0; j < 8; ++j) s0 += red[tid * 8 + j];
      rowv[tid] = 1.0f / s0;
    }
    __syncthreads();
  }

  // Normalize in LDS (ds_store only; coalesced tid -> m).
  for (int idx = tid; idx < 16 * SM; idx += 128) {
    const int r = idx >> 11;          // /2048
    const int m = idx & (SM - 1);
    smem[r * LDS_STRIDE + m] *= rowv[r];
  }
  __syncthreads();  // barrier waits out the ds_stores above for all waves

  // Stream align to HBM exactly once via the async DMA path:
  // each op copies 16B LDS -> global, tracked by ASYNCcnt.
  {
    float* aout = align + ((size_t)bh * SI + i0) * SM;
    const unsigned lbase = (unsigned)(size_t)(const void*)smem;  // LDS byte offset
    for (int idx = tid; idx < 16 * (SM / 4); idx += 128) {
      const int r = idx >> 9;          // 512 x 16B segments per row
      const int seg = idx & 511;
      const unsigned loff =
          lbase + (unsigned)(r * (LDS_STRIDE * 4) + seg * 16);
      const unsigned long long ga =
          (unsigned long long)(size_t)(aout + (size_t)r * SM + seg * 4);
      asm volatile("global_store_async_from_lds_b128 %0, %1, off"
                   :: "v"(ga), "v"(loff)
                   : "memory");
    }
    asm volatile("s_wait_asynccnt 0" ::: "memory");
  }

  // ---- Phase 3: ctx = align @ V (align from LDS, V transposed in HBM) -------
  {
    const int d0 = wave * 16;           // 4 waves cover Dh = 64
    v8f acc = {};
    for (int kk = 0; kk < SM; kk += 32) {
      // prefetch next V fragment line (global_prefetch_b8)
      if (kk + 32 < SM)
        __builtin_prefetch(Vt_ + (size_t)(d0 + (lane & 15)) * SM + kk + 32, 0, 0);
      v16bf a = load_A_frag_lds(smem, kk);
      v16bf bb = load_Bt_frag(Vt_, SM, d0, kk);
      acc = wmma_bf16(a, bb, acc);
    }
    const int dcol = h * DH + d0 + (lane & 15);
#pragma unroll
    for (int r = 0; r < 8; ++r) {
      const int i = i0 + r + 8 * half;
      ctx[((size_t)b * SI + i) * DD + dcol] = (bf16_t)acc[r];
    }
  }
}

// ---------------------------------------------------------------------------
// Launch
// ---------------------------------------------------------------------------
extern "C" void kernel_launch(void* const* d_in, const int* in_sizes, int n_in,
                              void* d_out, int out_size, void* d_ws, size_t ws_size,
                              hipStream_t stream) {
  (void)in_sizes; (void)n_in; (void)out_size; (void)ws_size;

  const float* in_x   = (const float*)d_in[0];  // [B,I,D]
  const float* in_mem = (const float*)d_in[1];  // [B,M,D]
  const float* bias   = (const float*)d_in[2];  // [B,I,M]
  const float* Wq     = (const float*)d_in[3];  // [D,D]
  const float* Wk     = (const float*)d_in[4];
  const float* Wv     = (const float*)d_in[5];
  const float* Wo     = (const float*)d_in[6];

  // Workspace carve (bf16 buffers, 256B aligned)
  char* ws = (char*)d_ws;
  size_t off = 0;
  auto carve = [&](size_t bytes) -> void* {
    void* p = ws + off;
    off += (bytes + 255) & ~(size_t)255;
    return p;
  };
  bf16_t* Xb  = (bf16_t*)carve((size_t)BB * SI * DD * 2);
  bf16_t* Mb  = (bf16_t*)carve((size_t)BB * SM * DD * 2);
  bf16_t* Wqt = (bf16_t*)carve((size_t)DD * DD * 2);
  bf16_t* Wkt = (bf16_t*)carve((size_t)DD * DD * 2);
  bf16_t* Wvt = (bf16_t*)carve((size_t)DD * DD * 2);
  bf16_t* Wot = (bf16_t*)carve((size_t)DD * DD * 2);
  bf16_t* Qb  = (bf16_t*)carve((size_t)BB * SI * DD * 2);
  bf16_t* Kb  = (bf16_t*)carve((size_t)BB * SM * DD * 2);
  bf16_t* Vt  = (bf16_t*)carve((size_t)BB * DD * SM * 2);
  bf16_t* ctx = (bf16_t*)carve((size_t)BB * SI * DD * 2);

  float* outp  = (float*)d_out;                      // [B,I,D]
  float* align = outp + (size_t)BB * SI * DD;        // [B,H,I,M]

  const float qscale = 0.125f;  // Dh^-0.5, Dh = 64

  // 1) casts (+ weight transpose, Q-scale folded into Wq)
  {
    int n = BB * SI * DD;
    cast_bf16<<<dim3((n + 255) / 256), dim3(256), 0, stream>>>(in_x, Xb, n);
  }
  {
    int n = BB * SM * DD;
    cast_bf16<<<dim3((n + 255) / 256), dim3(256), 0, stream>>>(in_mem, Mb, n);
  }
  {
    dim3 g((DD * DD + 255) / 256), blk(256);
    cast_w_trans<<<g, blk, 0, stream>>>(Wq, Wqt, qscale);
    cast_w_trans<<<g, blk, 0, stream>>>(Wk, Wkt, 1.0f);
    cast_w_trans<<<g, blk, 0, stream>>>(Wv, Wvt, 1.0f);
    cast_w_trans<<<g, blk, 0, stream>>>(Wo, Wot, 1.0f);
  }

  // 2) projections: rows = B*I (or B*M), cols = D, K = D
  {
    dim3 g(BB * SI / 16, DD / 256), blk(128);
    gemm_bf16<false, bf16_t><<<g, blk, 0, stream>>>(Xb, Wqt, Qb, DD, DD, DD, DD, 1, 1);
  }
  {
    dim3 g(BB * SM / 16, DD / 256), blk(128);
    gemm_bf16<false, bf16_t><<<g, blk, 0, stream>>>(Mb, Wkt, Kb, DD, DD, DD, DD, 1, 1);
    // V stored transposed: Vt[b, d, m]
    gemm_bf16<true, bf16_t><<<g, blk, 0, stream>>>(Mb, Wvt, Vt, DD, DD, DD, DD, SM, SM);
  }

  // 3) fused QK^T + bias + softmax + PV (align written to HBM exactly once,
  //    via async LDS->global DMA)
  {
    dim3 g(SI / 16, 1, BB * HH), blk(128);
    size_t shmem = (size_t)(16 * LDS_STRIDE + 128 + 16) * sizeof(float);
    fused_attn_kernel<<<g, blk, shmem, stream>>>(Qb, Kb, Vt, bias, align, ctx);
  }

  // 4) out = ctx @ Wo  (fp32 output)
  {
    dim3 g(BB * SI / 16, DD / 256), blk(128);
    gemm_bf16<false, float><<<g, blk, 0, stream>>>(ctx, Wot, outp, DD, DD, DD, DD, 1, 1);
  }
}
